// ARNet_40329742910151
// MI455X (gfx1250) — matrix operand order repaired and checked by reading, not compiled, not run on hardware
//
#include <hip/hip_runtime.h>
#include <math.h>

typedef __attribute__((ext_vector_type(16))) _Float16 v16h;
typedef __attribute__((ext_vector_type(8)))  float    v8f;

#define N_NODES 29
#define C_IN    6
#define K_NBR   6
#define D_FEAT  12      // 2*C
#define M_MSG   32
#define E_ROWS  174     // 29*6 edges
#define E_PAD   176     // pad to 11 tiles of 16
#define EIN     25
#define EIN_PAD 32
#define W1_N    50
#define W1_NPAD 64
#define NODE_K  44      // D + M
#define NODE_KPAD 64
#define NH1     24      // 2*D
#define NH1_PAD 32
#define H_HID   32

// HW reciprocal: single v_rcp_f32 instead of IEEE div chain.
__device__ __forceinline__ float sigmoidf_(float x) {
  return __builtin_amdgcn_rcpf(1.0f + __expf(-x));
}
__device__ __forceinline__ float siluf_(float x) { return x * sigmoidf_(x); }

// A fragment (16x32 f16), contiguous k order (k = 16*(lane/16) + e), matching the
// B packing. One contiguous 16-half (32B) chunk per lane -> 2x ds_load_b128.
__device__ __forceinline__ v16h load_a16(const _Float16* base, int ld, int row0, int lane) {
  const int m   = row0 + (lane & 15);
  const int khi = (lane >> 4) << 4;           // +16 for lanes 16..31
  v16h a;
#pragma unroll
  for (int e = 0; e < 16; ++e) {
    const int k = e + khi;
    a[e] = base[m * ld + k];
  }
  return a;
}

__global__ __launch_bounds__(128)
void arnet_egnn_fused(const float* __restrict__ x, const float* __restrict__ ctx,
                      const unsigned char* __restrict__ mask,
                      const float* __restrict__ e_w1, const float* __restrict__ e_b1,
                      const float* __restrict__ e_w2, const float* __restrict__ e_b2,
                      const float* __restrict__ g_w,  const float* __restrict__ g_b,
                      const float* __restrict__ n_w1, const float* __restrict__ n_b1,
                      const float* __restrict__ n_w2, const float* __restrict__ n_b2,
                      const float* __restrict__ h_w1, const float* __restrict__ h_b1,
                      const float* __restrict__ h_w2, const float* __restrict__ h_b2,
                      float* __restrict__ out)
{
  __shared__ float s_ctx[32][4];
  __shared__ float s_feats[32][16];           // zero-padded rows 29..31 and cols 12..15
  __shared__ float s_dist[32][32];
  __shared__ int   s_nbhd[32][8];
  __shared__ float s_rdist[32][8];
  __shared__ unsigned char s_mask[32];
  // Aliased buffer: edge-MLP input A1 [176x32] f16, then (barrier-separated) the
  // post-silu messages m_ij [176x32] f16.
  __shared__ _Float16 s_em[E_PAD][EIN_PAD];
  __shared__ _Float16 s_act1[E_PAD][W1_NPAD];
  // Weights pre-swizzled to WMMA B-fragment-major layout: [tile..][lane][16 halfs],
  // 32B contiguous per lane -> one aligned v16h load (2x ds_load_b128) per fragment.
  __shared__ __align__(32) _Float16 s_w1f[4][32][16];        // 4 n-tiles, K=32 step
  __shared__ __align__(32) _Float16 s_w2f[2][2][32][16];     // 2 n-tiles x 2 K-steps
  __shared__ __align__(32) _Float16 s_nw1f[2][2][32][16];    // 2 n-tiles x 2 K-steps
  __shared__ __align__(32) _Float16 s_nw2f[32][16];          // 1 tile
  __shared__ float    s_gate[E_PAD];
  __shared__ float    s_mi[32][M_MSG];
  __shared__ _Float16 s_nodeA[32][NODE_KPAD];
  __shared__ _Float16 s_h1[32][NH1_PAD];
  __shared__ float    s_nodeout[32][16];
  __shared__ float    s_eb1[W1_NPAD];         // zero-padded
  __shared__ float    s_eb2[M_MSG];
  __shared__ float    s_nb1[NH1_PAD];         // zero-padded
  __shared__ float    s_nb2[16];              // zero-padded
  __shared__ float    s_gw[M_MSG];
  __shared__ float    s_pool[16];
  __shared__ float    s_head[H_HID];

  const int t    = threadIdx.x;
  const int lane = t & 31;
  const int wave = t >> 5;
  const int b    = blockIdx.x;

  const float* xb = x   + (size_t)b * N_NODES * C_IN;
  const float* cb = ctx + (size_t)b * N_NODES * 3;
  float*       ob = out + (size_t)b * N_NODES * D_FEAT;

  if (t == 0) {
    __builtin_prefetch(e_w1); __builtin_prefetch(e_w2);
    __builtin_prefetch(n_w1); __builtin_prefetch(h_w2);
  }

  // ---- stage node data; feats = repeat(x,2), fully zero-padded [32][16] ----
  for (int i = t; i < 32 * 16; i += 128) {
    int n = i >> 4, c = i & 15;
    float v = 0.0f;
    if (n < N_NODES && c < D_FEAT) v = xb[n * C_IN + (c < C_IN ? c : c - C_IN)];
    s_feats[n][c] = v;
  }
  if (t < N_NODES) {
    s_mask[t] = mask[t];
    s_ctx[t][0] = cb[t*3+0]; s_ctx[t][1] = cb[t*3+1]; s_ctx[t][2] = cb[t*3+2];
  } else if (t < 32) {
    s_mask[t] = 0;
  }
  // ---- weights -> LDS f16, fragment-major, zero-padded ----
  for (int i = t; i < 4*32*16; i += 128) {          // e_w1 [25x50]
    int e = i & 15, ln = (i >> 4) & 31, nt = i >> 9;
    int n = nt*16 + (ln & 15);
    int k = ((ln >> 4) << 4) + e;
    s_w1f[nt][ln][e] = (_Float16)((k < EIN && n < W1_N) ? e_w1[k*W1_N + n] : 0.0f);
  }
  for (int i = t; i < 2*2*32*16; i += 128) {        // e_w2 [50x32]
    int e = i & 15, ln = (i >> 4) & 31, ks = (i >> 9) & 1, nt = i >> 10;
    int n = nt*16 + (ln & 15);
    int k = ks*32 + ((ln >> 4) << 4) + e;
    s_w2f[nt][ks][ln][e] = (_Float16)((k < W1_N) ? e_w2[k*M_MSG + n] : 0.0f);
  }
  for (int i = t; i < 2*2*32*16; i += 128) {        // n_w1 [44x24]
    int e = i & 15, ln = (i >> 4) & 31, ks = (i >> 9) & 1, nt = i >> 10;
    int n = nt*16 + (ln & 15);
    int k = ks*32 + ((ln >> 4) << 4) + e;
    s_nw1f[nt][ks][ln][e] = (_Float16)((k < NODE_K && n < NH1) ? n_w1[k*NH1 + n] : 0.0f);
  }
  for (int i = t; i < 32*16; i += 128) {            // n_w2 [24x12]
    int e = i & 15, ln = i >> 4;
    int n = ln & 15;
    int k = ((ln >> 4) << 4) + e;
    s_nw2f[ln][e] = (_Float16)((k < NH1 && n < D_FEAT) ? n_w2[k*D_FEAT + n] : 0.0f);
  }
  if (t < W1_NPAD) s_eb1[t] = (t < W1_N) ? e_b1[t] : 0.0f;
  if (t < M_MSG)   { s_eb2[t] = e_b2[t]; s_gw[t] = g_w[t]; }
  if (t < NH1_PAD) s_nb1[t] = (t < NH1) ? n_b1[t] : 0.0f;
  if (t < 16)      s_nb2[t] = (t < D_FEAT) ? n_b2[t] : 0.0f;
  for (int i = t; i < E_PAD*EIN_PAD; i += 128) (&s_em[0][0])[i] = (_Float16)0.0f;
  __syncthreads();

  // ---- pairwise squared distances ----
  for (int i = t; i < N_NODES*N_NODES; i += 128) {
    int r = i / N_NODES, c = i % N_NODES;
    float dx = s_ctx[r][0]-s_ctx[c][0];
    float dy = s_ctx[r][1]-s_ctx[c][1];
    float dz = s_ctx[r][2]-s_ctx[c][2];
    s_dist[r][c] = dx*dx + dy*dy + dz*dz;
  }
  __syncthreads();

  // ---- 6-NN, stable tie-break toward lowest index (matches lax.top_k) ----
  if (t < N_NODES) {
    unsigned chosen = 0;
    const bool mi = s_mask[t] != 0;
    for (int kk = 0; kk < K_NBR; ++kk) {
      float best = 3.4e38f; int bj = 0;
      for (int j = 0; j < N_NODES; ++j) {
        if (chosen & (1u << j)) continue;
        float r = (mi && s_mask[j]) ? s_dist[t][j] : 1e5f;
        if (r < best) { best = r; bj = j; }
      }
      chosen |= (1u << bj);
      s_nbhd[t][kk]  = bj;
      s_rdist[t][kk] = s_dist[t][bj];   // raw squared dist, as in the reference
    }
  }
  __syncthreads();

  // ---- edge features [feats_i | feats_j | rel_dist] -> f16 A matrix ----
  for (int e = t; e < E_ROWS; e += 128) {
    int n = e / K_NBR, kk = e % K_NBR;
    int j = s_nbhd[n][kk];
#pragma unroll
    for (int c = 0; c < D_FEAT; ++c) {
      s_em[e][c]        = (_Float16)s_feats[n][c];
      s_em[e][c+D_FEAT] = (_Float16)s_feats[j][c];
    }
    s_em[e][24] = (_Float16)s_rdist[n][kk];
  }
  __syncthreads();

  // ---- edge GEMM1: [176x25]@[25x50]; n-tile == wave, B fragment hoisted ----
  {
    const int nt = wave;
    const v16h w = *(const v16h*)&s_w1f[nt][lane][0];
    const int n = nt*16 + (lane & 15);
    const float bias = s_eb1[n];                 // zero on padded cols
    const int moff = (lane >> 4) << 3;
    for (int mt = 0; mt < 11; ++mt) {
      v16h a = load_a16(&s_em[0][0], EIN_PAD, mt*16, lane);
      v8f  c = {};
      c = __builtin_amdgcn_wmma_f32_16x16x32_f16(false, a, false, w, (short)0, c, false, false);
      const int mbase = mt*16 + moff;
#pragma unroll
      for (int i = 0; i < 8; ++i)
        s_act1[mbase + i][n] = (_Float16)siluf_(c[i] + bias);   // silu(0)=0 keeps pads zero
    }
  }
  __syncthreads();

  // ---- edge GEMM2: [176x50]@[50x32]; nt = wave&1, B fragments hoisted ----
  {
    const int nt = wave & 1;
    const v16h w0 = *(const v16h*)&s_w2f[nt][0][lane][0];
    const v16h w1 = *(const v16h*)&s_w2f[nt][1][lane][0];
    const int n = nt*16 + (lane & 15);
    const float bias = s_eb2[n];
    const int moff = (lane >> 4) << 3;
    for (int mt = (wave >> 1); mt < 11; mt += 2) {
      v8f c = {};
      v16h a0 = load_a16(&s_act1[0][0],  W1_NPAD, mt*16, lane);
      c = __builtin_amdgcn_wmma_f32_16x16x32_f16(false, a0, false, w0, (short)0, c, false, false);
      v16h a1 = load_a16(&s_act1[0][32], W1_NPAD, mt*16, lane);
      c = __builtin_amdgcn_wmma_f32_16x16x32_f16(false, a1, false, w1, (short)0, c, false, false);
      const int mbase = mt*16 + moff;
#pragma unroll
      for (int i = 0; i < 8; ++i)
        s_em[mbase + i][n] = (_Float16)siluf_(c[i] + bias);     // s_em now = m_ij
    }
  }
  __syncthreads();

  // ---- soft edge gate sigmoid(m @ g_w + g_b), fold in edge mask ----
  {
    const float gb = g_b[0];
    for (int e = t; e < E_ROWS; e += 128) {
      int n = e / K_NBR, kk = e % K_NBR;
      int j = s_nbhd[n][kk];
      const v16h m0 = *(const v16h*)&s_em[e][0];
      const v16h m1 = *(const v16h*)&s_em[e][16];
      float dot = gb;
#pragma unroll
      for (int c = 0; c < 16; ++c) dot += (float)m0[c] * s_gw[c];
#pragma unroll
      for (int c = 0; c < 16; ++c) dot += (float)m1[c] * s_gw[c+16];
      bool em = (s_mask[n] != 0) && (s_mask[j] != 0);
      s_gate[e] = em ? sigmoidf_(dot) : 0.0f;
    }
  }
  __syncthreads();

  // ---- m_i = sum over K neighbors of gated messages ----
  for (int i = t; i < N_NODES*M_MSG; i += 128) {
    int n = i / M_MSG, c = i % M_MSG;
    float acc = 0.0f;
#pragma unroll
    for (int kk = 0; kk < K_NBR; ++kk) {
      int e = n*K_NBR + kk;
      acc += (float)s_em[e][c] * s_gate[e];
    }
    s_mi[n][c] = acc;
  }
  __syncthreads();

  // ---- node_in = [feats | m_i], zero-padded ----
  for (int i = t; i < 32*NODE_KPAD; i += 128) {
    int n = i / NODE_KPAD, c = i % NODE_KPAD;
    float v = 0.0f;
    if (n < N_NODES) {
      if (c < D_FEAT)      v = s_feats[n][c];
      else if (c < NODE_K) v = s_mi[n][c - D_FEAT];
    }
    s_nodeA[n][c] = (_Float16)v;
  }
  __syncthreads();

  // ---- node MLP layer 1: [32x44]@[44x24]; one tile per wave, 2 K-steps ----
  {
    const int mt = wave >> 1, nt = wave & 1;
    const v16h w0 = *(const v16h*)&s_nw1f[nt][0][lane][0];
    const v16h w1 = *(const v16h*)&s_nw1f[nt][1][lane][0];
    v8f c = {};
    v16h a0 = load_a16(&s_nodeA[0][0],  NODE_KPAD, mt*16, lane);
    c = __builtin_amdgcn_wmma_f32_16x16x32_f16(false, a0, false, w0, (short)0, c, false, false);
    v16h a1 = load_a16(&s_nodeA[0][32], NODE_KPAD, mt*16, lane);
    c = __builtin_amdgcn_wmma_f32_16x16x32_f16(false, a1, false, w1, (short)0, c, false, false);
    const int n = nt*16 + (lane & 15);
    const float bias = s_nb1[n];
    const int mbase = mt*16 + ((lane >> 4) << 3);
#pragma unroll
    for (int i = 0; i < 8; ++i)
      s_h1[mbase + i][n] = (_Float16)siluf_(c[i] + bias);       // pads stay zero
  }
  __syncthreads();

  // ---- node MLP layer 2 + residual: [32x24]@[24x12] ----
  if (wave < 2) {
    const int mt = wave;
    const v16h w = *(const v16h*)&s_nw2f[lane][0];
    v16h a = load_a16(&s_h1[0][0], NH1_PAD, mt*16, lane);
    v8f  c = {};
    c = __builtin_amdgcn_wmma_f32_16x16x32_f16(false, a, false, w, (short)0, c, false, false);
    const int n = lane & 15;
    const float bias = s_nb2[n];
    const int mbase = mt*16 + ((lane >> 4) << 3);
#pragma unroll
    for (int i = 0; i < 8; ++i) {
      const int m = mbase + i;
      s_nodeout[m][n] = c[i] + bias + s_feats[m][n];            // unconditional, padded
    }
  }
  __syncthreads();

  // ---- masked mean pool over nodes ----
  if (t < D_FEAT) {
    float acc = 0.0f, cnt = 0.0f;
    for (int n = 0; n < N_NODES; ++n) {
      float mf = s_mask[n] ? 1.0f : 0.0f;
      acc += mf * s_nodeout[n][t];
      cnt += mf;
    }
    s_pool[t] = acc * __builtin_amdgcn_rcpf(cnt);
  }
  __syncthreads();

  // ---- head layer 1: relu(pooled @ h_w1 + h_b1), 12->32 ----
  if (t < H_HID) {
    float acc = h_b1[t];
#pragma unroll
    for (int i = 0; i < D_FEAT; ++i) acc += s_pool[i] * h_w1[i*H_HID + t];
    s_head[t] = fmaxf(acc, 0.0f);
  }
  __syncthreads();

  // ---- head layer 2 (32->24) into rows 0..1; zero-pad rows 2..28 ----
  for (int i = t; i < N_NODES*D_FEAT; i += 128) {
    float v = 0.0f;
    if (i < 2*D_FEAT) {
      v = h_b2[i];
#pragma unroll
      for (int k = 0; k < H_HID; ++k) v += s_head[k] * h_w2[k*(2*D_FEAT) + i];
    }
    ob[i] = v;
  }
}

extern "C" void kernel_launch(void* const* d_in, const int* in_sizes, int n_in,
                              void* d_out, int out_size, void* d_ws, size_t ws_size,
                              hipStream_t stream) {
  const float*         x    = (const float*)d_in[0];
  const float*         ctx  = (const float*)d_in[1];
  const unsigned char* mask = (const unsigned char*)d_in[2];
  const float* e_w1 = (const float*)d_in[3];
  const float* e_b1 = (const float*)d_in[4];
  const float* e_w2 = (const float*)d_in[5];
  const float* e_b2 = (const float*)d_in[6];
  const float* g_w  = (const float*)d_in[7];
  const float* g_b  = (const float*)d_in[8];
  const float* n_w1 = (const float*)d_in[9];
  const float* n_b1 = (const float*)d_in[10];
  const float* n_w2 = (const float*)d_in[11];
  const float* n_b2 = (const float*)d_in[12];
  const float* h_w1 = (const float*)d_in[13];
  const float* h_b1 = (const float*)d_in[14];
  const float* h_w2 = (const float*)d_in[15];
  const float* h_b2 = (const float*)d_in[16];
  float* out = (float*)d_out;

  const int Bcnt = in_sizes[0] / (N_NODES * C_IN);   // 8192
  arnet_egnn_fused<<<dim3(Bcnt), dim3(128), 0, stream>>>(
      x, ctx, mask, e_w1, e_b1, e_w2, e_b2, g_w, g_b,
      n_w1, n_b1, n_w2, n_b2, h_w1, h_b1, h_w2, h_b2, out);
}
